// MAXMNIST_73263552135860
// MI455X (gfx1250) — matrix operand order repaired and checked by reading, not compiled, not run on hardware
//
#include <hip/hip_runtime.h>
#include <hip/hip_bf16.h>

typedef __attribute__((ext_vector_type(16))) _Float16 v16h;
typedef __attribute__((ext_vector_type(8)))  float    v8f;

#define EPS_ACT 1e-5f
#define EPS_RMS 1e-6f
#define IPW 2            // images per workgroup
#define NT  256          // threads per block (8 waves of 32)

// ---- workspace layout (bytes) ----
#define WS_SCAL 0        // 8 floats: [0]=max|w1| [1]=max|w1b| [2]=alpha2 [3..6]=alpha fc1..fcl
#define WS_W1BQ 32       // 160*16 f16, quantized conv1b weights in WMMA B layout [K][oc]
#define WS_W2S  5152     // 96*144 i8 signs
#define WS_FC1S 18976    // 64*96 i8
#define WS_FC2S 25120    // 64*64 i8
#define WS_FC3S 29216    // 64*64 i8
#define WS_FCLS 33312    // 10*64 i8
#define WS_W1Q  33952    // 144 f32 quantized conv1 weights

__device__ __forceinline__ float qround(float x){ return rintf(x); }            // round-half-even like jnp.round
__device__ __forceinline__ float clip8(float x){ return fminf(fmaxf(x,-128.f),127.f); }

union U16 { v16h v; _Float16 e[16]; };
union U8F { v8f  v; float    e[8]; };

// ---------------- prep: per-tensor weight stats + quantized / sign tables ----------------
__global__ void __launch_bounds__(NT) bitnet_prep(
    const float* __restrict__ w1, const float* __restrict__ w1b,
    const float* __restrict__ w2, const float* __restrict__ f1,
    const float* __restrict__ f2, const float* __restrict__ f3,
    const float* __restrict__ fl, unsigned char* __restrict__ ws)
{
  __shared__ int   s_max1, s_max1b;
  __shared__ float s_sum[5];
  int tid = threadIdx.x;
  if (tid == 0) { s_max1 = 0; s_max1b = 0; }
  if (tid < 5)  s_sum[tid] = 0.f;
  __syncthreads();

  float m = 0.f;
  for (int i = tid; i < 144;  i += NT) m = fmaxf(m, fabsf(w1[i]));
  atomicMax(&s_max1,  __float_as_int(m));            // bits of non-negative floats are order-preserving
  m = 0.f;
  for (int i = tid; i < 2304; i += NT) m = fmaxf(m, fabsf(w1b[i]));
  atomicMax(&s_max1b, __float_as_int(m));
  float s = 0.f;
  for (int i = tid; i < 96*144; i += NT) s += fabsf(w2[i]);
  atomicAdd(&s_sum[0], s);
  s = 0.f; for (int i = tid; i < 64*96; i += NT) s += fabsf(f1[i]); atomicAdd(&s_sum[1], s);
  s = 0.f; for (int i = tid; i < 64*64; i += NT) s += fabsf(f2[i]); atomicAdd(&s_sum[2], s);
  s = 0.f; for (int i = tid; i < 64*64; i += NT) s += fabsf(f3[i]); atomicAdd(&s_sum[3], s);
  s = 0.f; for (int i = tid; i < 10*64; i += NT) s += fabsf(fl[i]); atomicAdd(&s_sum[4], s);
  __syncthreads();

  float* scal = (float*)(ws + WS_SCAL);
  if (tid == 0) {
    scal[0] = __int_as_float(s_max1);
    scal[1] = __int_as_float(s_max1b);
    scal[2] = s_sum[0] / (96.f*144.f);
    scal[3] = s_sum[1] / (64.f*96.f);
    scal[4] = s_sum[2] / 4096.f;
    scal[5] = s_sum[3] / 4096.f;
    scal[6] = s_sum[4] / 640.f;
  }

  // quantized conv1 weights (8-bit, per-tensor)
  float sw1 = 127.f / fmaxf(__int_as_float(s_max1), EPS_ACT);
  float* w1q = (float*)(ws + WS_W1Q);
  for (int i = tid; i < 144; i += NT) w1q[i] = clip8(qround(w1[i]*sw1)) / sw1;

  // quantized conv1b weights (8-bit, per-tensor) in WMMA B layout [K=160][oc=16], K zero-padded
  float sw1b = 127.f / fmaxf(__int_as_float(s_max1b), EPS_ACT);
  _Float16* w1bq = (_Float16*)(ws + WS_W1BQ);
  for (int idx = tid; idx < 160*16; idx += NT) {
    int K = idx >> 4, oc = idx & 15;
    float q = 0.f;
    if (K < 144) {
      int ic = K/9, r = K - ic*9, kh = r/3, kw = r - kh*3;
      float w = w1b[oc*144 + ic*9 + kh*3 + kw];
      q = clip8(qround(w*sw1b)) / sw1b;
    }
    w1bq[idx] = (_Float16)q;
  }

  // sign tables for binary tensors
  signed char* w2s  = (signed char*)(ws + WS_W2S);
  signed char* f1s  = (signed char*)(ws + WS_FC1S);
  signed char* f2s  = (signed char*)(ws + WS_FC2S);
  signed char* f3s  = (signed char*)(ws + WS_FC3S);
  signed char* fls  = (signed char*)(ws + WS_FCLS);
  for (int i = tid; i < 96*144; i += NT) w2s[i] = (signed char)((w2[i]>0.f)?1:((w2[i]<0.f)?-1:0));
  for (int i = tid; i < 64*96;  i += NT) f1s[i] = (signed char)((f1[i]>0.f)?1:((f1[i]<0.f)?-1:0));
  for (int i = tid; i < 64*64;  i += NT) f2s[i] = (signed char)((f2[i]>0.f)?1:((f2[i]<0.f)?-1:0));
  for (int i = tid; i < 64*64;  i += NT) f3s[i] = (signed char)((f3[i]>0.f)?1:((f3[i]<0.f)?-1:0));
  for (int i = tid; i < 10*64;  i += NT) fls[i] = (signed char)((fl[i]>0.f)?1:((fl[i]<0.f)?-1:0));
}

// ---------------- fused bit_linear helper (RMSNorm -> act quant -> binary matmul) ----------------
__device__ __forceinline__ void bit_linear(
    const float* sin, int inStride, int Nin,
    float* qbuf,                     // [IPW*96] scratch
    const signed char* wsgn, float alpha,
    float* sout, int outStride, int Nout, bool relu,
    float* sums, int* mxs, int tid,
    float* gout, int goutStride)
{
  if (tid < IPW) { sums[tid] = 0.f; mxs[tid] = 0; }
  __syncthreads();
  for (int i = tid; i < IPW*Nin; i += NT) {
    int img = i / Nin, k = i - img*Nin;
    float v = sin[img*inStride + k];
    atomicAdd(&sums[img], v*v);
  }
  __syncthreads();
  for (int i = tid; i < IPW*Nin; i += NT) {
    int img = i / Nin, k = i - img*Nin;
    float rms = rsqrtf(sums[img] / (float)Nin + EPS_RMS);
    float xn  = sin[img*inStride + k] * rms;
    atomicMax(&mxs[img], __float_as_int(fabsf(xn)));
  }
  __syncthreads();
  for (int i = tid; i < IPW*Nin; i += NT) {
    int img = i / Nin, k = i - img*Nin;
    float rms = rsqrtf(sums[img] / (float)Nin + EPS_RMS);
    float sc  = 127.f / fmaxf(__int_as_float(mxs[img]), EPS_ACT);
    float xn  = sin[img*inStride + k] * rms;
    qbuf[img*96 + k] = clip8(qround(xn*sc)) / sc;
  }
  __syncthreads();
  for (int o = tid; o < IPW*Nout; o += NT) {
    int img = o / Nout, j = o - img*Nout;
    float acc = 0.f;
    for (int k = 0; k < Nin; k++)
      acc += qbuf[img*96 + k] * (float)wsgn[j*Nin + k];
    float y = acc * alpha;
    if (relu) y = fmaxf(y, 0.f);
    if (gout) gout[img*goutStride + j] = y;
    else      sout[img*outStride + j]  = y;
  }
  __syncthreads();
}

// ---------------- fused forward pass, IPW images per workgroup ----------------
__global__ void __launch_bounds__(NT) bitnet_fused(
    const float* __restrict__ x, const unsigned char* __restrict__ ws,
    float* __restrict__ out)
{
  __shared__ float      s_x[IPW][16][16];
  __shared__ float      s_w1q[144];
  __shared__ _Float16   s_w1bq[160][16];
  __shared__ _Float16   s_haq[IPW][16][14][14];
  __shared__ float      s_h2[IPW][16][12][12];
  __shared__ float      s_scale2[IPW][16][12];
  __shared__ signed char s_w2s[96*144];
  __shared__ float      s_h3[IPW][96];
  __shared__ float      s_xqn[IPW*96];
  __shared__ float      s_hfc[IPW][64];
  __shared__ float      s_sum[IPW];
  __shared__ int        s_mx[IPW];

  int tid  = threadIdx.x;
  int lane = tid & 31;
  int wid  = tid >> 5;
  int imgBase = blockIdx.x * IPW;

  const float* scal = (const float*)(ws + WS_SCAL);
  float alpha2 = scal[2], afc1 = scal[3], afc2 = scal[4], afc3 = scal[5], afcl = scal[6];

  // stage in inputs + weights
  for (int i = tid; i < IPW*256; i += NT) {
    int img = i >> 8, p = i & 255;
    s_x[img][p >> 4][p & 15] = x[(size_t)(imgBase + img)*256 + p];
  }
  for (int i = tid; i < 144;  i += NT) s_w1q[i] = ((const float*)(ws + WS_W1Q))[i];
  for (int i = tid; i < 2560; i += NT) ((_Float16*)s_w1bq)[i] = ((const _Float16*)(ws + WS_W1BQ))[i];
  for (int i = tid; i < 96*144; i += NT) s_w2s[i] = ((const signed char*)(ws + WS_W2S))[i];
  __syncthreads();

  // act-quant x: per (img,row) of width 16, in place
  for (int r = tid; r < IPW*16; r += NT) {
    int img = r >> 4, h = r & 15;
    float mx = 0.f;
    #pragma unroll
    for (int w = 0; w < 16; w++) mx = fmaxf(mx, fabsf(s_x[img][h][w]));
    float sc = 127.f / fmaxf(mx, EPS_ACT);
    #pragma unroll
    for (int w = 0; w < 16; w++) s_x[img][h][w] = clip8(qround(s_x[img][h][w]*sc)) / sc;
  }
  __syncthreads();

  // conv1 (1->16ch, 3x3) + ReLU + per-row act-quant -> f16 A tensor
  for (int r = tid; r < IPW*16*14; r += NT) {
    int img = r / 224, rem = r - img*224, oc = rem / 14, oh = rem % 14;
    float vals[14]; float mx = 0.f;
    #pragma unroll
    for (int ow = 0; ow < 14; ow++) {
      float acc = 0.f;
      #pragma unroll
      for (int kh = 0; kh < 3; kh++)
        #pragma unroll
        for (int kw = 0; kw < 3; kw++)
          acc += s_x[img][oh+kh][ow+kw] * s_w1q[oc*9 + kh*3 + kw];
      acc = fmaxf(acc, 0.f);
      vals[ow] = acc; mx = fmaxf(mx, acc);
    }
    float sc = 127.f / fmaxf(mx, EPS_ACT);
    #pragma unroll
    for (int ow = 0; ow < 14; ow++)
      s_haq[img][oc][oh][ow] = (_Float16)(clip8(qround(vals[ow]*sc)) / sc);
  }
  __syncthreads();

  // ---- conv1b as implicit GEMM on v_wmma_f32_16x16x32_f16 ----
  // D[16 pix x 16 oc] = A[16 pix x 160 K] * B[160 K x 16 oc], K=(ic,kh,kw) padded 144->160
  U16 bfrag[5];
  #pragma unroll
  for (int c = 0; c < 5; c++)
    #pragma unroll
    for (int i = 0; i < 16; i++)           // B layout: K = c*32 + 16*(lane>=16) + i, N = lane&15
      bfrag[c].e[i] = s_w1bq[c*32 + ((lane >> 4) << 4) + i][lane & 15];

  for (int job = wid; job < IPW*9; job += 8) {     // wave-uniform job loop, EXEC all ones
    int img = job / 9, mt = job - img*9;
    int p  = mt*16 + (lane & 15);                  // A: M = lane&15
    int oh = p / 12, ow = p - oh*12;
    U8F acc;
    #pragma unroll
    for (int i = 0; i < 8; i++) acc.e[i] = 0.f;
    #pragma unroll
    for (int c = 0; c < 5; c++) {
      U16 a;
      #pragma unroll
      for (int i = 0; i < 16; i++) {               // A layout: K = i + 16*(lane>=16)
        int K = c*32 + i + ((lane >> 4) << 4);
        _Float16 av = (_Float16)0.f;
        if (K < 144) {
          int ic = K/9, r2 = K - ic*9, kh = r2/3, kw = r2 - kh*3;
          av = s_haq[img][ic][oh+kh][ow+kw];
        }
        a.e[i] = av;
      }
      acc.v = __builtin_amdgcn_wmma_f32_16x16x32_f16(
          false, a.v, false, bfrag[c].v, (short)0, acc.v, false, false);
    }
    int oc = lane & 15;                            // D layout: M = vreg + 8*(lane>=16), N = lane&15
    #pragma unroll
    for (int vv = 0; vv < 8; vv++) {
      int pp = mt*16 + vv + ((lane >> 4) << 3);
      int oh2 = pp / 12, ow2 = pp - oh2*12;
      s_h2[img][oc][oh2][ow2] = fmaxf(acc.e[vv], 0.f);
    }
  }
  __syncthreads();

  // per-row act scales for h2 (rows of width 12, values >= 0 after ReLU)
  for (int r = tid; r < IPW*16*12; r += NT) {
    int img = r / 192, rem = r - img*192, c = rem / 12, h = rem % 12;
    float mx = 0.f;
    #pragma unroll
    for (int w = 0; w < 12; w++) mx = fmaxf(mx, s_h2[img][c][h][w]);
    s_scale2[img][c][h] = 127.f / fmaxf(mx, EPS_ACT);
  }
  __syncthreads();

  // conv2: grouped (16 groups, 1 ic -> 6 oc), binary weights, full 12x12 window, + ReLU
  for (int o = tid; o < IPW*96; o += NT) {
    int img = o / 96, oc = o - img*96, ic = oc / 6;
    float acc = 0.f;
    for (int kh = 0; kh < 12; kh++) {
      float sc = s_scale2[img][ic][kh], inv = 1.f / sc;
      #pragma unroll
      for (int kw = 0; kw < 12; kw++) {
        float xq = fminf(qround(s_h2[img][ic][kh][kw]*sc), 127.f) * inv;
        acc += xq * (float)s_w2s[oc*144 + kh*12 + kw];
      }
    }
    s_h3[img][oc] = fmaxf(acc * alpha2, 0.f);
  }
  __syncthreads();

  // FC chain: RMSNorm + act quant + binary matmul
  bit_linear(&s_h3[0][0], 96, 96, s_xqn, (const signed char*)(ws + WS_FC1S), afc1,
             &s_hfc[0][0], 64, 64, true,  s_sum, s_mx, tid, nullptr, 0);
  bit_linear(&s_hfc[0][0], 64, 64, s_xqn, (const signed char*)(ws + WS_FC2S), afc2,
             &s_hfc[0][0], 64, 64, true,  s_sum, s_mx, tid, nullptr, 0);
  bit_linear(&s_hfc[0][0], 64, 64, s_xqn, (const signed char*)(ws + WS_FC3S), afc3,
             &s_hfc[0][0], 64, 64, true,  s_sum, s_mx, tid, nullptr, 0);
  bit_linear(&s_hfc[0][0], 64, 64, s_xqn, (const signed char*)(ws + WS_FCLS), afcl,
             nullptr, 0, 10, false, s_sum, s_mx, tid,
             out + (size_t)imgBase*10, 10);
}

extern "C" void kernel_launch(void* const* d_in, const int* in_sizes, int n_in,
                              void* d_out, int out_size, void* d_ws, size_t ws_size,
                              hipStream_t stream) {
  (void)n_in; (void)out_size; (void)ws_size;
  const float* x   = (const float*)d_in[0];
  const float* w1  = (const float*)d_in[1];
  const float* w1b = (const float*)d_in[2];
  const float* w2  = (const float*)d_in[3];
  const float* f1  = (const float*)d_in[4];
  const float* f2  = (const float*)d_in[5];
  const float* f3  = (const float*)d_in[6];
  const float* fl  = (const float*)d_in[7];
  unsigned char* ws = (unsigned char*)d_ws;
  int B = in_sizes[0] / 256;                 // x is [B,1,16,16]

  bitnet_prep<<<1, NT, 0, stream>>>(w1, w1b, w2, f1, f2, f3, fl, ws);
  bitnet_fused<<<B / IPW, NT, 0, stream>>>(x, ws, (float*)d_out);
}
